// VanillaGCN_34918084116972
// MI455X (gfx1250) — matrix operand order repaired and checked by reading, not compiled, not run on hardware
//
#include <hip/hip_runtime.h>
#include <hip/hip_bf16.h>

typedef __attribute__((ext_vector_type(16))) __bf16 v16bf;
typedef __attribute__((ext_vector_type(8)))  __bf16 v8bf;
typedef __attribute__((ext_vector_type(8)))  float  v8f;

__device__ inline unsigned short f2bf_bits(float f) {
    unsigned int u = __float_as_uint(f);
    unsigned int r = u + 0x7FFFu + ((u >> 16) & 1u);   // round-to-nearest-even
    return (unsigned short)(r >> 16);
}

// ---------------------------------------------------------------------------
// Weight f32 -> bf16, swizzled to WMMA-B fragment-major order:
//   out[(kc*4 + nt)*512 + lane*16 + i]  =  W[k, n]   (zero-padded for k >= din)
//   k = kc*32 + (lane>>4)*16 + i,  n = nt*16 + (lane&15)
// so each lane's 16 B-fragment elements are 32 contiguous bytes.
// ---------------------------------------------------------------------------
__global__ void wconv_kernel(const float* __restrict__ W, unsigned short* __restrict__ out,
                             int din, int Ktot) {
    int idx = blockIdx.x * 256 + threadIdx.x;
    if (idx >= Ktot * 64) return;
    int kc   = idx >> 11;
    int rem  = idx & 2047;
    int nt   = rem >> 9;
    int lane = (rem >> 4) & 31;
    int i    = rem & 15;
    int k = kc * 32 + (lane >> 4) * 16 + i;
    int n = nt * 16 + (lane & 15);
    float v = (k < din) ? W[k * 64 + n] : 0.f;
    out[idx] = f2bf_bits(v);
}

__global__ void zero_kernel(float* __restrict__ p, long long n) {
    long long i = (long long)blockIdx.x * 256 + threadIdx.x;
    if (i < n) p[i] = 0.f;
}

// ---------------------------------------------------------------------------
// Bidirectional scatter-add: msg[end] += h[start]; msg[start] += h[end]
// 16 threads/edge, 4 floats/thread, hardware f32 atomics; h is L2-resident.
// ---------------------------------------------------------------------------
__global__ __launch_bounds__(256) void scatter_kernel(
    const float* __restrict__ h, const int* __restrict__ srt, const int* __restrict__ dst,
    float* __restrict__ msg, int E)
{
    long long t = (long long)blockIdx.x * 256 + threadIdx.x;
    if (t >= (long long)E * 16) return;
    int e = (int)(t >> 4);
    int c = (int)(t & 15) * 4;
    int s = srt[e], d = dst[e];
    const float4 hs = *(const float4*)&h[(size_t)s * 64 + c];
    const float4 hd = *(const float4*)&h[(size_t)d * 64 + c];
    float* md = &msg[(size_t)d * 64 + c];
    float* ms = &msg[(size_t)s * 64 + c];
    unsafeAtomicAdd(&md[0], hs.x); unsafeAtomicAdd(&md[1], hs.y);
    unsafeAtomicAdd(&md[2], hs.z); unsafeAtomicAdd(&md[3], hs.w);
    unsafeAtomicAdd(&ms[0], hd.x); unsafeAtomicAdd(&ms[1], hd.y);
    unsafeAtomicAdd(&ms[2], hd.z); unsafeAtomicAdd(&ms[3], hd.w);
}

// ---------------------------------------------------------------------------
// Fused Linear(+bias) -> LayerNorm -> ReLU via V_WMMA_F32_16X16X32_BF16.
// A row r = [in0[src0(r)][0:Kin0] | in1[src1(r)][0:Kin1] | 0-pad to KTOT].
// 256 thr = 8 waves; wave w owns rows 16w..16w+15 x all 64 cols (4 acc tiles).
// KTOT compile-time -> K-loop fully unrolled (4*KTOT/32 WMMAs in flight).
// ---------------------------------------------------------------------------
template <int KTOT>
__global__ __launch_bounds__(256) void mlp_layer_kernel(
    const float* __restrict__ in0, int Kin0,
    const float* __restrict__ in1, int Kin1,
    const int* __restrict__ idx0, const int* __restrict__ idx1,
    const unsigned short* __restrict__ Wq,
    const float* __restrict__ bias, const float* __restrict__ gamma,
    const float* __restrict__ beta, float* __restrict__ out, int rows)
{
    __shared__ __align__(32) unsigned short sW[KTOT * 64];  // swizzled bf16 weights
    __shared__ __align__(16) unsigned short sA[128 * 32];   // bf16 A chunk [128 rows][32 k]
    __shared__ float sOut[128 * 65];                        // f32 outputs, padded stride 65
    __shared__ float sBias[64], sG[64], sBe[64];
    __shared__ int sI0[128], sI1[128];

    const int tid   = threadIdx.x;
    const int lane  = tid & 31;
    const int wave  = tid >> 5;
    const int mlane = lane & 15;
    const int hi    = lane >> 4;
    const int rowBase = blockIdx.x * 128;

    // coalesced weight copy (layout already fragment-major)
    {
        const uint32_t* src = (const uint32_t*)Wq;
        uint32_t* dstw = (uint32_t*)sW;
#pragma unroll
        for (int i = tid; i < KTOT * 32; i += 256) dstw[i] = src[i];
    }
    if (tid < 64) { sBias[tid] = bias[tid]; sG[tid] = gamma[tid]; sBe[tid] = beta[tid]; }
    if (tid < 128) {
        int gr = rowBase + tid;
        sI0[tid] = (idx0 && gr < rows) ? idx0[gr] : gr;
        sI1[tid] = (idx1 && gr < rows) ? idx1[gr] : gr;
    }

    v8f acc[4];
#pragma unroll
    for (int i = 0; i < 4; i++) { v8f z = {}; acc[i] = z; }

#pragma unroll
    for (int kc = 0; kc < (KTOT >> 5); kc++) {
        __syncthreads();
        // stage A chunk: float4 global loads -> bf16 ushort4 LDS stores
        for (int j = tid; j < 128 * 8; j += 256) {
            int r = j >> 3, q = (j & 7) * 4;
            int gr = rowBase + r;
            int k = (kc << 5) + q;
            float4 v = make_float4(0.f, 0.f, 0.f, 0.f);
            if (gr < rows) {
                if (k + 3 < Kin0) {
                    v = *(const float4*)&in0[(size_t)sI0[r] * Kin0 + k];
                } else if (k >= Kin0 && k + 3 < Kin0 + Kin1) {
                    v = *(const float4*)&in1[(size_t)sI1[r] * Kin1 + (k - Kin0)];
                } else {
                    float e[4];
#pragma unroll
                    for (int t = 0; t < 4; t++) {
                        int kk = k + t; float s = 0.f;
                        if (kk < Kin0)             s = in0[(size_t)sI0[r] * Kin0 + kk];
                        else if (kk < Kin0 + Kin1) s = in1[(size_t)sI1[r] * Kin1 + (kk - Kin0)];
                        e[t] = s;
                    }
                    v = make_float4(e[0], e[1], e[2], e[3]);
                }
            }
            ushort4 pk;
            pk.x = f2bf_bits(v.x); pk.y = f2bf_bits(v.y);
            pk.z = f2bf_bits(v.z); pk.w = f2bf_bits(v.w);
            *(ushort4*)&sA[r * 32 + q] = pk;
        }
        __syncthreads();

        // A fragment: two aligned 16B LDS loads + shuffle
        const unsigned short* ar = &sA[(wave * 16 + mlane) * 32];
        const int kb = hi * 16;
        v8bf a0 = *(const v8bf*)(ar + kb);
        v8bf a1 = *(const v8bf*)(ar + kb + 8);
        v16bf a = __builtin_shufflevector(a0, a1, 0, 1, 2, 3, 4, 5, 6, 7,
                                                  8, 9, 10, 11, 12, 13, 14, 15);
#pragma unroll
        for (int nt = 0; nt < 4; nt++) {
            // B fragment: one aligned 32B LDS read (fragment-major layout)
            v16bf b = *(const v16bf*)&sW[(kc * 4 + nt) * 512 + lane * 16];
            acc[nt] = __builtin_amdgcn_wmma_f32_16x16x32_bf16(
                false, a, false, b, (short)0, acc[nt], false, false);
        }
    }

    // spill accumulators (+linear bias): lane holds (M = v + 8*hi, N = nt*16 + mlane)
#pragma unroll
    for (int nt = 0; nt < 4; nt++)
#pragma unroll
        for (int v = 0; v < 8; v++) {
            int row = wave * 16 + v + hi * 8;
            int col = nt * 16 + mlane;
            sOut[row * 65 + col] = acc[nt][v] + sBias[col];
        }
    __syncthreads();

    // LayerNorm + ReLU epilogue, one thread per row
    if (tid < 128) {
        int gr = rowBase + tid;
        const float* rp = &sOut[tid * 65];
        float m = 0.f;
#pragma unroll
        for (int c = 0; c < 64; c++) m += rp[c];
        m *= (1.f / 64.f);
        float var = 0.f;
#pragma unroll
        for (int c = 0; c < 64; c++) { float d = rp[c] - m; var += d * d; }
        var *= (1.f / 64.f);
        float inv = rsqrtf(var + 1e-5f);
        if (gr < rows) {
#pragma unroll
            for (int c = 0; c < 64; c++) {
                float y = (rp[c] - m) * inv * sG[c] + sBe[c];
                out[(size_t)gr * 64 + c] = y > 0.f ? y : 0.f;
            }
        }
    }
}

// ---------------------------------------------------------------------------
// Final edge head: out[r] = dot(ehid[r], Wf) + bf
// ---------------------------------------------------------------------------
__global__ __launch_bounds__(256) void edge_head_kernel(
    const float* __restrict__ eh, const float* __restrict__ Wf,
    const float* __restrict__ bf, float* __restrict__ out, int E)
{
    __shared__ float sw[64];
    if (threadIdx.x < 64) sw[threadIdx.x] = Wf[threadIdx.x];
    __syncthreads();
    int r = blockIdx.x * 256 + threadIdx.x;
    if (r >= E) return;
    const float* rp = &eh[(size_t)r * 64];
    float acc = bf[0];
#pragma unroll
    for (int k = 0; k < 64; k++) acc += rp[k] * sw[k];
    out[r] = acc;
}

// ---------------------------------------------------------------------------
extern "C" void kernel_launch(void* const* d_in, const int* in_sizes, int n_in,
                              void* d_out, int out_size, void* d_ws, size_t ws_size,
                              hipStream_t stream) {
    const float* x  = (const float*)d_in[0];
    const int*   ei = (const int*)d_in[1];
    const int N = in_sizes[0] / 3;
    const int E = in_sizes[1] / 2;
    const int* startI = ei;
    const int* endI   = ei + E;

    // input dict order: x, edge_index, enc{W*3,b*3,g*3,beta*3}, node{...}, edge{hid{...}, Wf, bf}
    const float* Wsrc[9] = {
        (const float*)d_in[2],  (const float*)d_in[3],  (const float*)d_in[4],    // enc W
        (const float*)d_in[14], (const float*)d_in[15], (const float*)d_in[16],   // node W
        (const float*)d_in[26], (const float*)d_in[27], (const float*)d_in[28] }; // edge-hid W
    const float* Bv[9] = { (const float*)d_in[5],  (const float*)d_in[6],  (const float*)d_in[7],
                           (const float*)d_in[17], (const float*)d_in[18], (const float*)d_in[19],
                           (const float*)d_in[29], (const float*)d_in[30], (const float*)d_in[31] };
    const float* Gv[9] = { (const float*)d_in[8],  (const float*)d_in[9],  (const float*)d_in[10],
                           (const float*)d_in[20], (const float*)d_in[21], (const float*)d_in[22],
                           (const float*)d_in[32], (const float*)d_in[33], (const float*)d_in[34] };
    const float* Bt[9] = { (const float*)d_in[11], (const float*)d_in[12], (const float*)d_in[13],
                           (const float*)d_in[23], (const float*)d_in[24], (const float*)d_in[25],
                           (const float*)d_in[35], (const float*)d_in[36], (const float*)d_in[37] };
    const float* Wf = (const float*)d_in[38];
    const float* bf = (const float*)d_in[39];

    const int din [9] = { 3, 64, 64, 128, 64, 64, 128, 64, 64 };
    const int ktot[9] = { 32, 64, 64, 128, 64, 64, 128, 64, 64 };

    // workspace carve-up
    char* p = (char*)d_ws;
    auto alloc = [&](size_t bytes) { void* r = (void*)p; p += (bytes + 255) & ~(size_t)255; return r; };
    const int ECH = 1 << 18;                 // edge chunk (256k rows) to bound workspace
    float* h   = (float*)alloc((size_t)N * 64 * 4);
    float* tmp = (float*)alloc((size_t)N * 64 * 4);
    float* msg = (float*)alloc((size_t)N * 64 * 4);
    float* e0  = (float*)alloc((size_t)ECH * 64 * 4);
    float* e1  = (float*)alloc((size_t)ECH * 64 * 4);
    unsigned short* Wq[9];
    for (int i = 0; i < 9; i++) Wq[i] = (unsigned short*)alloc((size_t)ktot[i] * 64 * 2);

    for (int i = 0; i < 9; i++)
        wconv_kernel<<<(ktot[i] * 64 + 255) / 256, 256, 0, stream>>>(Wsrc[i], Wq[i], din[i], ktot[i]);

    auto mlp = [&](const float* i0, int K0, const float* i1, int K1,
                   const int* g0, const int* g1, int w, float* outp, int rows) {
        int grid = (rows + 127) / 128;
        switch (ktot[w]) {
        case 32:
            mlp_layer_kernel<32><<<grid, 256, 0, stream>>>(
                i0, K0, i1, K1, g0, g1, Wq[w], Bv[w], Gv[w], Bt[w], outp, rows);
            break;
        case 64:
            mlp_layer_kernel<64><<<grid, 256, 0, stream>>>(
                i0, K0, i1, K1, g0, g1, Wq[w], Bv[w], Gv[w], Bt[w], outp, rows);
            break;
        default:
            mlp_layer_kernel<128><<<grid, 256, 0, stream>>>(
                i0, K0, i1, K1, g0, g1, Wq[w], Bv[w], Gv[w], Bt[w], outp, rows);
            break;
        }
    };

    // node encoder: x -> msg -> tmp -> h
    mlp(x,   3,  nullptr, 0, nullptr, nullptr, 0, msg, N);
    mlp(msg, 64, nullptr, 0, nullptr, nullptr, 1, tmp, N);
    mlp(tmp, 64, nullptr, 0, nullptr, nullptr, 2, h,   N);

    // 8 message-passing iterations
    for (int it = 0; it < 8; it++) {
        long long nz = (long long)N * 64;
        zero_kernel<<<(unsigned)((nz + 255) / 256), 256, 0, stream>>>(msg, nz);
        long long st = (long long)E * 16;
        scatter_kernel<<<(unsigned)((st + 255) / 256), 256, 0, stream>>>(h, startI, endI, msg, E);
        mlp(h,   64, msg, 64, nullptr, nullptr, 3, tmp, N);   // [h|msg] -> tmp
        mlp(tmp, 64, nullptr, 0, nullptr, nullptr, 4, msg, N);
        mlp(msg, 64, nullptr, 0, nullptr, nullptr, 5, h,   N);
    }

    // edge head, chunked: concat(h[start], h[end]) -> MLP -> linear
    for (int c0 = 0; c0 < E; c0 += ECH) {
        int C = (E - c0 < ECH) ? (E - c0) : ECH;
        mlp(h,  64, h, 64, startI + c0, endI + c0, 6, e0, C);
        mlp(e0, 64, nullptr, 0, nullptr, nullptr, 7, e1, C);
        mlp(e1, 64, nullptr, 0, nullptr, nullptr, 8, e0, C);
        edge_head_kernel<<<(C + 255) / 256, 256, 0, stream>>>(e0, Wf, bf, (float*)d_out + c0, C);
    }
}
